// Generator_29807073034794
// MI455X (gfx1250) — compile-verified, matrix operand order checked by reading, not run on hardware
//
#include <hip/hip_runtime.h>

typedef _Float16 v16h __attribute__((ext_vector_type(16)));
typedef float    v8f  __attribute__((ext_vector_type(8)));
typedef float    f4_t __attribute__((ext_vector_type(4)));
typedef _Float16 h4_t __attribute__((ext_vector_type(4)));
typedef int      v4i  __attribute__((ext_vector_type(4)));

static constexpr int CB = 32;    // batch
static constexpr int CT = 24;    // timesteps
static constexpr int CN = 1024;  // nodes
static constexpr int CU = 16;    // rnn units

union U16 { v16h v; f4_t f4[2]; };

__device__ __forceinline__ float sigf(float x) { return 1.f / (1.f + __expf(-x)); }

#if defined(__has_builtin)
#if __has_builtin(__builtin_amdgcn_global_load_async_to_lds_b128) && \
    __has_builtin(__builtin_amdgcn_s_wait_asynccnt)
#define HAS_ASYNC_LDS 1
#endif
#endif

// ---------------- A (f32) -> A (f16), 32M elements, vectorized ----------------
__global__ void k_cvtA(const float* __restrict__ A, _Float16* __restrict__ Ah, int n4) {
    for (int i = blockIdx.x * blockDim.x + threadIdx.x; i < n4; i += gridDim.x * blockDim.x) {
        f4_t x = ((const f4_t*)A)[i];
        h4_t y;
        y[0] = (_Float16)x[0]; y[1] = (_Float16)x[1];
        y[2] = (_Float16)x[2]; y[3] = (_Float16)x[3];
        ((h4_t*)Ah)[i] = y;
    }
}

// ---------------- big batched GEMM: Y = A(1024x1024,f16) @ Zt + fused epilogue ----
// Zt layout: (B, KC, N) f16 (contiguous K per column). Whole Z panel is staged in
// LDS via GLOBAL_LOAD_ASYNC_TO_LDS_B128 (XOR-swizzled per column to spread banks);
// each wave computes one 16-row tile x ALL column tiles, reusing each A fragment
// across NCT WMMAs, with the next A fragment preloaded during the current WMMA.
// MODE 0: RU  (KC=32, 2 col tiles): r=sig(y+rb), u=sig(y+ub)
// MODE 1: C   (KC=16): h = u*h + (1-u)*tanh(y+cb)
// MODE 2: FIN (KC=16): out[...,0:2] = tanh(y+gcn_b)
template <int MODE>
__global__ __launch_bounds__(256) void k_gemm(
    const _Float16* __restrict__ Ah, const _Float16* __restrict__ Zt,
    float* __restrict__ out0, float* __restrict__ out1,
    const float* __restrict__ bias0, const float* __restrict__ bias1,
    const float* __restrict__ uvec) {
    constexpr int NCT = (MODE == 0) ? 2 : 1;
    constexpr int KC  = (MODE == 0) ? 32 : 1 * 16;
    __shared__ _Float16 sB[KC * CN];  // 64 KB (RU) / 32 KB (C,FIN) of 320 KB WGP LDS

    const int lane = threadIdx.x & 31;
    const int wv   = threadIdx.x >> 5;
    const int rb_  = blockIdx.x & 7;
    const int b    = blockIdx.x >> 3;
    const int mBase = (rb_ * 8 + wv) * 16;
    const int off   = (lane & 16) ? 8 : 0;   // K sub-offset per ISA 16-bit fragment layout
    const int arow  = mBase + (lane & 15);
    const int colL  = lane & 15;

    // ---- stage full Z panel for this batch into LDS (swizzled) ----
    for (int c = threadIdx.x; c < KC * (CN / 8); c += 256) {
        int col  = c >> 7;                 // CN/8 == 128 chunks per column
        int koff = (c & 127) << 3;         // 8-half (16B) chunks
        int dsto = (col << 10) + (koff ^ ((col & 15) << 3));
        const _Float16* src = Zt + (((size_t)b * KC + col) << 10) + koff;
#if HAS_ASYNC_LDS
        __builtin_amdgcn_global_load_async_to_lds_b128(
            (__attribute__((address_space(1))) v4i*)(void*)src,
            (__attribute__((address_space(3))) v4i*)(void*)&sB[dsto], 0, 0);
#else
        *(f4_t*)&sB[dsto] = *(const f4_t*)src;
#endif
    }
#if HAS_ASYNC_LDS
    __builtin_amdgcn_s_wait_asynccnt(0);
#endif
    __syncthreads();

    const _Float16* pA = Ah + ((size_t)b << 20) + ((size_t)arow << 10) + off;

    v8f acc[NCT];
#pragma unroll
    for (int ct = 0; ct < NCT; ct++) acc[ct] = (v8f){};

    U16 a_cur, a_nxt;
    a_cur.f4[0] = *(const f4_t*)(pA);
    a_cur.f4[1] = *(const f4_t*)(pA + 16);

    const int swz = colL << 3;
#pragma unroll
    for (int k0 = 0; k0 < CN; k0 += 32) {
        __builtin_prefetch(pA + k0 + 512, 0, 1);  // unconditional, no branch
        if (k0 + 32 < CN) {
            a_nxt.f4[0] = *(const f4_t*)(pA + k0 + 32);
            a_nxt.f4[1] = *(const f4_t*)(pA + k0 + 48);
        }
#pragma unroll
        for (int ct = 0; ct < NCT; ct++) {
            const int cbase = (ct * 16 + colL) << 10;
            U16 bm;
            bm.f4[0] = *(const f4_t*)&sB[cbase + ((k0 + off) ^ swz)];
            bm.f4[1] = *(const f4_t*)&sB[cbase + ((k0 + off + 16) ^ swz)];
            acc[ct] = __builtin_amdgcn_wmma_f32_16x16x32_f16(
                false, a_cur.v, false, bm.v, (short)0, acc[ct], false, false);
        }
        a_cur = a_nxt;
    }

    const int row0 = mBase + ((lane >> 4) << 3);
    if (MODE == 0) {
#pragma unroll
        for (int ct = 0; ct < NCT; ct++) {
            float bs   = (ct == 0 ? bias0 : bias1)[colL];
            float* dst = (ct == 0 ? out0 : out1);
#pragma unroll
            for (int i = 0; i < 8; i++) {
                size_t idx = ((((size_t)b << 10) + row0 + i) << 4) + colL;
                dst[idx] = sigf(acc[ct][i] + bs);
            }
        }
    } else if (MODE == 1) {
        float bs = bias0[colL];
#pragma unroll
        for (int i = 0; i < 8; i++) {
            size_t idx = ((((size_t)b << 10) + row0 + i) << 4) + colL;
            float uu = uvec[idx], hh = out0[idx];
            float cc = tanhf(acc[0][i] + bs);
            out0[idx] = uu * hh + (1.f - uu) * cc;
        }
    } else {
        if (colL < 2) {
            float bs = bias0[colL];
#pragma unroll
            for (int i = 0; i < 8; i++) {
                size_t idx = ((((size_t)b << 10) + row0 + i) << 1) + colL;
                out0[idx] = tanhf(acc[0][i] + bs);
            }
        }
    }
}

// ---------------- per-node transforms (tiny), writing transposed Z (B, cols, N) ----
__global__ void k_zru1(const float* __restrict__ recent, const float* __restrict__ h1,
                       const float* __restrict__ rW, const float* __restrict__ uW,
                       int t, _Float16* __restrict__ Zt) {
    int tid = blockIdx.x * blockDim.x + threadIdx.x;  // b*1024+n
    int b = tid >> 10, n = tid & 1023;
    const float* xp = recent + ((((size_t)b * CT + t) << 10) + n) * 2;
    float x0 = xp[0], x1 = xp[1];
    float h[16];
#pragma unroll
    for (int k = 0; k < 16; k++) h[k] = h1[((size_t)tid << 4) + k];
    for (int j = 0; j < 32; j++) {
        const float* w = (j < 16) ? (rW + j * 18) : (uW + (j - 16) * 18);
        float a = w[0] * x0 + w[1] * x1;
#pragma unroll
        for (int k = 0; k < 16; k++) a += w[2 + k] * h[k];
        Zt[(((size_t)b << 5) + j) * CN + n] = (_Float16)a;
    }
}

__global__ void k_zc1(const float* __restrict__ recent, const float* __restrict__ h1,
                      const float* __restrict__ r, const float* __restrict__ cW,
                      int t, _Float16* __restrict__ Zt) {
    int tid = blockIdx.x * blockDim.x + threadIdx.x;
    int b = tid >> 10, n = tid & 1023;
    const float* xp = recent + ((((size_t)b * CT + t) << 10) + n) * 2;
    float x0 = xp[0], x1 = xp[1];
    float rh[16];
#pragma unroll
    for (int k = 0; k < 16; k++) rh[k] = r[((size_t)tid << 4) + k] * h1[((size_t)tid << 4) + k];
    for (int j = 0; j < 16; j++) {
        const float* w = cW + j * 18;
        float a = w[0] * x0 + w[1] * x1;
#pragma unroll
        for (int k = 0; k < 16; k++) a += w[2 + k] * rh[k];
        Zt[(((size_t)b << 4) + j) * CN + n] = (_Float16)a;
    }
}

__global__ void k_zru2(const float* __restrict__ h1, const float* __restrict__ h2,
                       const float* __restrict__ rW, const float* __restrict__ uW,
                       _Float16* __restrict__ Zt) {
    int tid = blockIdx.x * blockDim.x + threadIdx.x;
    int b = tid >> 10, n = tid & 1023;
    float a1[16], a2[16];
#pragma unroll
    for (int k = 0; k < 16; k++) { a1[k] = h1[((size_t)tid << 4) + k]; a2[k] = h2[((size_t)tid << 4) + k]; }
    for (int j = 0; j < 32; j++) {
        const float* w = (j < 16) ? (rW + j * 32) : (uW + (j - 16) * 32);
        float a = 0.f;
#pragma unroll
        for (int k = 0; k < 16; k++) a += w[k] * a1[k] + w[16 + k] * a2[k];
        Zt[(((size_t)b << 5) + j) * CN + n] = (_Float16)a;
    }
}

__global__ void k_zc2(const float* __restrict__ h1, const float* __restrict__ h2,
                      const float* __restrict__ r, const float* __restrict__ cW,
                      _Float16* __restrict__ Zt) {
    int tid = blockIdx.x * blockDim.x + threadIdx.x;
    int b = tid >> 10, n = tid & 1023;
    float a1[16], rh[16];
#pragma unroll
    for (int k = 0; k < 16; k++) {
        a1[k] = h1[((size_t)tid << 4) + k];
        rh[k] = r[((size_t)tid << 4) + k] * h2[((size_t)tid << 4) + k];
    }
    for (int j = 0; j < 16; j++) {
        const float* w = cW + j * 32;
        float a = 0.f;
#pragma unroll
        for (int k = 0; k < 16; k++) a += w[k] * a1[k] + w[16 + k] * rh[k];
        Zt[(((size_t)b << 4) + j) * CN + n] = (_Float16)a;
    }
}

__global__ void k_zfinal(const float* __restrict__ h2, const float* __restrict__ trendH,
                         const float* __restrict__ feat, const float* __restrict__ gcnW,
                         _Float16* __restrict__ Zt) {
    int tid = blockIdx.x * blockDim.x + threadIdx.x;
    int b = tid >> 10, n = tid & 1023;
    float h[16];
#pragma unroll
    for (int k = 0; k < 16; k++) h[k] = h2[((size_t)tid << 4) + k];
    for (int c = 0; c < 16; c++) {
        float a = 0.f;
        if (c < 2) {
            const float* w = gcnW + c * 80;
#pragma unroll
            for (int k = 0; k < 16; k++) a += h[k] * w[k];
            for (int k = 0; k < 32; k++) a += trendH[b * 32 + k] * w[16 + k];
            for (int k = 0; k < 32; k++) a += feat[b * 32 + k] * w[48 + k];
        }
        Zt[(((size_t)b << 4) + c) * CN + n] = (_Float16)a;
    }
}

// ---------------- 2-layer LSTM over trend + FF branch (tiny, one block) ----------
__global__ __launch_bounds__(1024) void k_lstm_ff(
    const float* __restrict__ trend, const float* __restrict__ tf,
    const float* __restrict__ Wih0, const float* __restrict__ Whh0,
    const float* __restrict__ bih0, const float* __restrict__ bhh0,
    const float* __restrict__ Wih1, const float* __restrict__ Whh1,
    const float* __restrict__ bih1, const float* __restrict__ bhh1,
    const float* __restrict__ ffW, const float* __restrict__ ffb,
    float* __restrict__ hs0, float* __restrict__ trendH, float* __restrict__ feat) {
    __shared__ float sh[CB * 32], sc[CB * 32];
    int tid = threadIdx.x, b = tid >> 5, j = tid & 31;
    sh[tid] = 0.f; sc[tid] = 0.f;
    __syncthreads();
    for (int t = 0; t < CT; t++) {
        float x0 = trend[(b * CT + t) * 2 + 0], x1 = trend[(b * CT + t) * 2 + 1];
        float g[4];
        for (int gg = 0; gg < 4; gg++) {
            int row = gg * 32 + j;
            float a = Wih0[row * 2] * x0 + Wih0[row * 2 + 1] * x1 + bih0[row] + bhh0[row];
            for (int k = 0; k < 32; k++) a += sh[b * 32 + k] * Whh0[row * 32 + k];
            g[gg] = a;
        }
        __syncthreads();
        float cc = sigf(g[1]) * sc[tid] + sigf(g[0]) * tanhf(g[2]);
        float hh = sigf(g[3]) * tanhf(cc);
        sc[tid] = cc; sh[tid] = hh;
        hs0[(t * CB + b) * 32 + j] = hh;
        __syncthreads();
    }
    sh[tid] = 0.f; sc[tid] = 0.f;
    __syncthreads();
    for (int t = 0; t < CT; t++) {
        float g[4];
        for (int gg = 0; gg < 4; gg++) {
            int row = gg * 32 + j;
            float a = bih1[row] + bhh1[row];
            for (int k = 0; k < 32; k++) a += hs0[(t * CB + b) * 32 + k] * Wih1[row * 32 + k];
            for (int k = 0; k < 32; k++) a += sh[b * 32 + k] * Whh1[row * 32 + k];
            g[gg] = a;
        }
        __syncthreads();
        float cc = sigf(g[1]) * sc[tid] + sigf(g[0]) * tanhf(g[2]);
        float hh = sigf(g[3]) * tanhf(cc);
        sc[tid] = cc; sh[tid] = hh;
        __syncthreads();
    }
    trendH[tid] = sh[tid];
    float a = ffb[j];
    for (int k = 0; k < 31; k++) a += tf[b * 31 + k] * ffW[j * 31 + k];
    feat[tid] = fmaxf(a, 0.f);
}

extern "C" void kernel_launch(void* const* d_in, const int* in_sizes, int n_in,
                              void* d_out, int out_size, void* d_ws, size_t ws_size,
                              hipStream_t stream) {
    const float* recent = (const float*)d_in[0];
    const float* trend  = (const float*)d_in[1];
    const float* A      = (const float*)d_in[2];
    const float* tf     = (const float*)d_in[3];
    const float* g1rW = (const float*)d_in[4];  const float* g1rb = (const float*)d_in[5];
    const float* g1uW = (const float*)d_in[6];  const float* g1ub = (const float*)d_in[7];
    const float* g1cW = (const float*)d_in[8];  const float* g1cb = (const float*)d_in[9];
    const float* g2rW = (const float*)d_in[10]; const float* g2rb = (const float*)d_in[11];
    const float* g2uW = (const float*)d_in[12]; const float* g2ub = (const float*)d_in[13];
    const float* g2cW = (const float*)d_in[14]; const float* g2cb = (const float*)d_in[15];
    const float* Wih0 = (const float*)d_in[16]; const float* Whh0 = (const float*)d_in[17];
    const float* bih0 = (const float*)d_in[18]; const float* bhh0 = (const float*)d_in[19];
    const float* Wih1 = (const float*)d_in[20]; const float* Whh1 = (const float*)d_in[21];
    const float* bih1 = (const float*)d_in[22]; const float* bhh1 = (const float*)d_in[23];
    const float* ffW  = (const float*)d_in[24]; const float* ffb  = (const float*)d_in[25];
    const float* gcnW = (const float*)d_in[26]; const float* gcnb = (const float*)d_in[27];
    float* out = (float*)d_out;

    char* ws = (char*)d_ws;
    _Float16* Ah = (_Float16*)ws;                                  // 64 MB
    float* h1 = (float*)(ws + ((size_t)64 << 20));                 // 2 MB each
    float* h2 = h1 + (size_t)CB * CN * CU;
    float* rr = h2 + (size_t)CB * CN * CU;
    float* uu = rr + (size_t)CB * CN * CU;
    _Float16* Zru = (_Float16*)(uu + (size_t)CB * CN * CU);        // (B,32,N) f16
    _Float16* Zc  = Zru + (size_t)CB * 32 * CN;                    // (B,16,N) f16
    float* hs0    = (float*)(Zc + (size_t)CB * 16 * CN);           // (T,B,32)
    float* trendH = hs0 + (size_t)CT * CB * 32;
    float* feat   = trendH + (size_t)CB * 32;
    _Float16* Zf  = (_Float16*)(feat + (size_t)CB * 32);           // (B,16,N) f16

    // 1) A -> fp16 (64 MB, L2-resident thereafter)
    k_cvtA<<<4096, 256, 0, stream>>>(A, Ah, (CB * CN * CN) / 4);
    // 2) zero initial hidden states (h1, h2 contiguous)
    (void)hipMemsetAsync(h1, 0, (size_t)2 * CB * CN * CU * sizeof(float), stream);
    // 3) LSTM + FF branch (independent of GRU loop)
    k_lstm_ff<<<1, 1024, 0, stream>>>(trend, tf, Wih0, Whh0, bih0, bhh0,
                                      Wih1, Whh1, bih1, bhh1, ffW, ffb,
                                      hs0, trendH, feat);

    const int SM_GRID = (CB * CN) / 256;  // 128
    const int GG = CB * 8;                // 256 blocks, 8 waves each
    for (int t = 0; t < CT; t++) {
        // ---- GRU layer 1 ----
        k_zru1<<<SM_GRID, 256, 0, stream>>>(recent, h1, g1rW, g1uW, t, Zru);
        k_gemm<0><<<GG, 256, 0, stream>>>(Ah, Zru, rr, uu, g1rb, g1ub, nullptr);
        k_zc1<<<SM_GRID, 256, 0, stream>>>(recent, h1, rr, g1cW, t, Zc);
        k_gemm<1><<<GG, 256, 0, stream>>>(Ah, Zc, h1, nullptr, g1cb, nullptr, uu);
        // ---- GRU layer 2 ----
        k_zru2<<<SM_GRID, 256, 0, stream>>>(h1, h2, g2rW, g2uW, Zru);
        k_gemm<0><<<GG, 256, 0, stream>>>(Ah, Zru, rr, uu, g2rb, g2ub, nullptr);
        k_zc2<<<SM_GRID, 256, 0, stream>>>(h1, h2, rr, g2cW, Zc);
        k_gemm<1><<<GG, 256, 0, stream>>>(Ah, Zc, h2, nullptr, g2cb, nullptr, uu);
    }

    // final GCN: combined @ gcn_W^T folded per-node, then A @ Zf with tanh epilogue
    k_zfinal<<<SM_GRID, 256, 0, stream>>>(h2, trendH, feat, gcnW, Zf);
    k_gemm<2><<<GG, 256, 0, stream>>>(Ah, Zf, out, nullptr, gcnb, nullptr, nullptr);
}
